// SparseStaticGraphReservoir_7249904796085
// MI455X (gfx1250) — matrix operand order repaired and checked by reading, not compiled, not run on hardware
//
#include <hip/hip_runtime.h>
#include <math.h>

// ---------------------------------------------------------------------------
// Problem constants (match reference)
// ---------------------------------------------------------------------------
#define N_NODES 40000
#define N_EDGES 640000
#define HID     128
#define N_ITERS 8   // after the initial state = tanh(x_proj) update (total 9)

typedef __attribute__((ext_vector_type(16))) __bf16 v16bf;
typedef __attribute__((ext_vector_type(8)))  __bf16 v8bf;
typedef __attribute__((ext_vector_type(8)))  float  v8f;

// Packed bf16 atomic add (2 x bf16 per op) — CDNA5 GLOBAL_ATOMIC_PK_ADD_BF16,
// non-returning form (TH=0), via inline asm (portable across toolchains).
__device__ __forceinline__ void pk_add_bf16(unsigned short* p, unsigned int v) {
    asm volatile("global_atomic_pk_add_bf16 %0, %1, off"
                 :: "v"(p), "v"(v) : "memory");
}

// ---------------------------------------------------------------------------
// Pre-swizzle the 4 weight matrices into WMMA B-fragment layout (bf16).
// frag[mat][ntile][kstep][lane][j]:
//   n = ntile*16 + (lane & 15)
//   k = kstep*32 + (j < 8 ? (lane >> 4)*8 + j : 16 + (lane >> 4)*8 + (j - 8))
// mats 0/2 (w_in0/w_in1) used transposed: B[k][n] = W[n*128 + k]
// mats 1/3 (w_rec0/w_rec1):              B[k][n] = W[k*128 + n]
// ---------------------------------------------------------------------------
__global__ __launch_bounds__(256)
void prep_frags_kernel(const float* __restrict__ w_in0,
                       const float* __restrict__ w_rec0,
                       const float* __restrict__ w_in1,
                       const float* __restrict__ w_rec1,
                       __bf16* __restrict__ frag)
{
    int idx   = blockIdx.x * 256 + threadIdx.x;      // 0 .. 65535
    int j     = idx & 15;
    int lane  = (idx >> 4) & 31;
    int kstep = (idx >> 9) & 3;
    int ntile = (idx >> 11) & 7;
    int mat   = idx >> 14;

    int n = ntile * 16 + (lane & 15);
    int k = kstep * 32 + (j < 8 ? (lane >> 4) * 8 + j : 16 + (lane >> 4) * 8 + (j - 8));

    float v;
    if (mat == 0)       v = w_in0 [n * HID + k];   // transposed
    else if (mat == 1)  v = w_rec0[k * HID + n];
    else if (mat == 2)  v = w_in1 [n * HID + k];   // transposed
    else                v = w_rec1[k * HID + n];

    frag[idx] = (__bf16)v;
}

// ---------------------------------------------------------------------------
// f32 -> bf16 elementwise convert (2 elems / thread)
// ---------------------------------------------------------------------------
__global__ __launch_bounds__(256)
void f32_to_bf16_kernel(const float2* __restrict__ in, __bf16* __restrict__ out)
{
    int i = blockIdx.x * 256 + threadIdx.x;
    float2 v = in[i];
    out[2 * i + 0] = (__bf16)v.x;
    out[2 * i + 1] = (__bf16)v.y;
}

// state = tanh(xproj), bf16 out (first fixed-point update: state_0 == 0)
__global__ __launch_bounds__(256)
void tanh_init_kernel(const float2* __restrict__ xproj, __bf16* __restrict__ state)
{
    int i = blockIdx.x * 256 + threadIdx.x;
    float2 v = xproj[i];
    state[2 * i + 0] = (__bf16)tanhf(v.x);
    state[2 * i + 1] = (__bf16)tanhf(v.y);
}

// zero bf16 buffer, 16 bytes / thread
__global__ __launch_bounds__(256)
void zero_kernel(uint4* __restrict__ p)
{
    int i = blockIdx.x * 256 + threadIdx.x;
    p[i] = make_uint4(0u, 0u, 0u, 0u);
}

// ---------------------------------------------------------------------------
// Fused GEMM:  acc = A(bf16, 40000x128) @ Bfrag(bf16, 128x128), f32 accum.
// MODE 0: out_f = acc                    (projection -> x_proj)
// MODE 1: out_b = bf16(tanh(xproj+acc))  (state update)
// MODE 2: out_f = tanh(xproj+acc)        (final state -> d_out, f32)
// Block = 256 threads = 8 waves; block owns a 16-row M tile, wave w owns
// N tile w; K = 128 -> 4 chained v_wmma_f32_16x16x32_bf16.
// A-fragment (16-bit A layout): lane half h, row = mbase+(lane&15):
//   elems 0..7  = K k0 + 8h + 0..7      (contiguous 16B)
//   elems 8..15 = K k0 + 16 + 8h + 0..7 (contiguous 16B)
// C/D: col = nbase + (lane&15); VGPR r -> row = mbase + 8h + r
// ---------------------------------------------------------------------------
template <int MODE>
__global__ __launch_bounds__(256)
void gemm_kernel(const __bf16* __restrict__ A,
                 const __bf16* __restrict__ bfrag,   // [8][4][32][16]
                 const float*  __restrict__ xproj,   // MODE 1,2
                 float*        __restrict__ out_f,   // MODE 0,2
                 __bf16*       __restrict__ out_b)   // MODE 1
{
    const int lane  = threadIdx.x & 31;
    const int wave  = threadIdx.x >> 5;
    const int half  = lane >> 4;
    const int l15   = lane & 15;
    const int mbase = blockIdx.x * 16;
    const int nbase = wave * 16;

    const __bf16* arow = A + (mbase + l15) * HID + half * 8;

    v8f c = {};
#pragma unroll
    for (int kk = 0; kk < 4; ++kk) {
        const int k0 = kk * 32;
        v8bf lo = *(const v8bf*)(arow + k0);
        v8bf hi = *(const v8bf*)(arow + k0 + 16);
        v16bf a = __builtin_shufflevector(lo, hi,
                    0, 1, 2, 3, 4, 5, 6, 7, 8, 9, 10, 11, 12, 13, 14, 15);

        v16bf b = *(const v16bf*)(bfrag + ((wave * 4 + kk) * 32 + lane) * 16);

        c = __builtin_amdgcn_wmma_f32_16x16x32_bf16(
                /*neg_a=*/false, a, /*neg_b=*/false, b,
                /*c_mod=*/(short)0, c, /*reuse_a=*/false, /*reuse_b=*/false);
    }

    const int col = nbase + l15;
#pragma unroll
    for (int r = 0; r < 8; ++r) {
        int row = mbase + half * 8 + r;
        int idx = row * HID + col;
        if (MODE == 0) {
            out_f[idx] = c[r];
        } else {
            float v = tanhf(xproj[idx] + c[r]);
            if (MODE == 1) out_b[idx] = (__bf16)v;
            else           out_f[idx] = v;
        }
    }
}

// ---------------------------------------------------------------------------
// Edge scatter: aggr[dst[e]] += state[src[e]]  (128 bf16 per edge)
// 16 lanes per edge: one uint4 (8 bf16) gather + 4x packed bf16 atomic add.
// ---------------------------------------------------------------------------
__global__ __launch_bounds__(256)
void scatter_kernel(const __bf16* __restrict__ state,
                    const int*    __restrict__ src,
                    const int*    __restrict__ dst,
                    __bf16*       __restrict__ aggr)
{
    int t = blockIdx.x * 256 + threadIdx.x;
    int e = t >> 4;               // 16 edges per block
    int c = (t & 15) << 3;        // bf16 feature base (8 per lane)
    int s = src[e];
    int d = dst[e];
    uint4 v = *(const uint4*)(state + s * HID + c);
    unsigned short* p = (unsigned short*)(aggr + d * HID + c);
    pk_add_bf16(p + 0, v.x);
    pk_add_bf16(p + 2, v.y);
    pk_add_bf16(p + 4, v.z);
    pk_add_bf16(p + 6, v.w);
}

// ---------------------------------------------------------------------------
// Launch
// ---------------------------------------------------------------------------
extern "C" void kernel_launch(void* const* d_in, const int* in_sizes, int n_in,
                              void* d_out, int out_size, void* d_ws, size_t ws_size,
                              hipStream_t stream)
{
    (void)in_sizes; (void)n_in; (void)out_size; (void)ws_size;

    const int*   edge_index = (const int*)  d_in[0];   // [2, N_EDGES]
    const float* x          = (const float*)d_in[1];   // [N, 128] f32
    const float* w_in0      = (const float*)d_in[2];
    const float* w_rec0     = (const float*)d_in[3];
    const float* w_in1      = (const float*)d_in[4];
    const float* w_rec1     = (const float*)d_in[5];
    float*       out        = (float*)d_out;

    const int* src = edge_index;
    const int* dst = edge_index + N_EDGES;

    // workspace carve
    const size_t MATSZ = (size_t)N_NODES * HID;        // 5,120,000 elems
    char* ws = (char*)d_ws;
    float*  xproj    = (float*) (ws);                              // 20.48 MB
    __bf16* state_bf = (__bf16*)(ws + MATSZ * 4);                  // 10.24 MB
    __bf16* aggr_bf  = (__bf16*)(ws + MATSZ * 4 + MATSZ * 2);      // 10.24 MB
    __bf16* x_bf     = (__bf16*)(ws + MATSZ * 4 + MATSZ * 4);      // 10.24 MB
    __bf16* frag     = (__bf16*)(ws + MATSZ * 4 + MATSZ * 6);      // 128 KB
    __bf16* frag_win0  = frag;
    __bf16* frag_wrec0 = frag + 16384;
    __bf16* frag_win1  = frag + 2 * 16384;
    __bf16* frag_wrec1 = frag + 3 * 16384;

    const dim3 blk(256);
    const dim3 grid_gemm(N_NODES / 16);            // 2500
    const dim3 grid_edge(N_EDGES / 16);            // 40000 (16 lanes/edge)
    const dim3 grid_pair(MATSZ / 2 / 256);         // 10000 (2 elems/thread)
    const dim3 grid_zero(MATSZ * 2 / 16 / 256);    // 2500  (16 B/thread)

    // 0) weight fragments + x -> bf16
    prep_frags_kernel<<<dim3(65536 / 256), blk, 0, stream>>>(
        w_in0, w_rec0, w_in1, w_rec1, frag);
    f32_to_bf16_kernel<<<grid_pair, blk, 0, stream>>>((const float2*)x, x_bf);

    // ---------------- layer 0 ----------------
    // x_proj = x @ w_in0^T
    gemm_kernel<0><<<grid_gemm, blk, 0, stream>>>(x_bf, frag_win0, nullptr, xproj, nullptr);
    // state_1 = tanh(x_proj)
    tanh_init_kernel<<<grid_pair, blk, 0, stream>>>((const float2*)xproj, state_bf);
    for (int it = 0; it < N_ITERS; ++it) {
        zero_kernel<<<grid_zero, blk, 0, stream>>>((uint4*)aggr_bf);
        scatter_kernel<<<grid_edge, blk, 0, stream>>>(state_bf, src, dst, aggr_bf);
        gemm_kernel<1><<<grid_gemm, blk, 0, stream>>>(aggr_bf, frag_wrec0, xproj, nullptr, state_bf);
    }

    // ---------------- layer 1 ----------------
    // x_proj = h0 @ w_in1^T   (h0 currently in state_bf)
    gemm_kernel<0><<<grid_gemm, blk, 0, stream>>>(state_bf, frag_win1, nullptr, xproj, nullptr);
    tanh_init_kernel<<<grid_pair, blk, 0, stream>>>((const float2*)xproj, state_bf);
    for (int it = 0; it < N_ITERS; ++it) {
        zero_kernel<<<grid_zero, blk, 0, stream>>>((uint4*)aggr_bf);
        scatter_kernel<<<grid_edge, blk, 0, stream>>>(state_bf, src, dst, aggr_bf);
        if (it == N_ITERS - 1)
            gemm_kernel<2><<<grid_gemm, blk, 0, stream>>>(aggr_bf, frag_wrec1, xproj, out, nullptr);
        else
            gemm_kernel<1><<<grid_gemm, blk, 0, stream>>>(aggr_bf, frag_wrec1, xproj, nullptr, state_bf);
    }
}
